// MoeLayerTopPLayer_63496796504070
// MI455X (gfx1250) — compile-verified, matrix operand order checked by reading, not run on hardware
//
#include <hip/hip_runtime.h>
#include <math.h>
#include <stdint.h>

// ---- problem constants ----
#define Tn 65536
#define Dn 256
#define En 8
#define Hn 512
#define On 256
#define TOPP 0.7f

// ---- ffn tiling ----
#define MT   32      // tokens per block tile
#define XS_S 264     // xs row stride (256 + 8 pad, 16B-aligned rows)
#define W1_S 264     // w1^T row stride (K=256 + 8)
#define W2_S 72      // w2^T row stride (K=64 + 8)
#define HS_S 72      // h row stride (64 + 8)

typedef __attribute__((ext_vector_type(16))) __bf16 v16bf;
typedef __attribute__((ext_vector_type(8)))  __bf16 v8bf;
typedef __attribute__((ext_vector_type(8)))  float  v8f;

// ---- CDNA5 async global->LDS copy (ASYNCcnt-tracked), ISA ch.10 / 08_async_tensor ----
__device__ __forceinline__ void async_copy_b128(uint32_t lds_off, const void* gaddr) {
    asm volatile("global_load_async_to_lds_b128 %0, %1, off"
                 :: "v"(lds_off), "v"(gaddr) : "memory");
}
__device__ __forceinline__ void wait_asynccnt0() {
    asm volatile("s_wait_asynccnt 0x0" ::: "memory");
}

// Load one 16x16x32 bf16 A/B fragment from a row-major LDS tile (ISA 7.12.2 layout):
// half-wave 0 holds K {0..7,16..23}, half-wave 1 holds {8..15,24..31} -> two 16B reads.
__device__ __forceinline__ v16bf ldfrag(const __bf16* p) {
    v8bf lo = *(const v8bf*)(p);
    v8bf hi = *(const v8bf*)(p + 16);
    v16bf f;
#pragma unroll
    for (int i = 0; i < 8; ++i) { f[i] = lo[i]; f[i + 8] = hi[i]; }
    return f;
}

// Branch-free erf (Abramowitz-Stegun 7.1.26, |err| <= 1.5e-7): rcp + exp + 5 FMA,
// no EXEC divergence (libm erff's range-split branches serialize the wave).
__device__ __forceinline__ float erf_fast(float x) {
    const float ax = fabsf(x);
    const float t  = __builtin_amdgcn_rcpf(fmaf(0.3275911f, ax, 1.0f));
    float poly = fmaf(1.061405429f, t, -1.453152027f);
    poly = fmaf(poly, t,  1.421413741f);
    poly = fmaf(poly, t, -0.284496736f);
    poly = fmaf(poly, t,  0.254829592f);
    poly *= t;
    const float r = 1.0f - poly * __expf(-ax * ax);
    return copysignf(r, x);
}
__device__ __forceinline__ float gelu_fast(float v) {
    return 0.5f * v * (1.0f + erf_fast(v * 0.70710678118654752440f));
}

// ---------------- tiny init: zero per-expert counters ----------------
__global__ void init_counts_kernel(int* __restrict__ counts) {
    if (threadIdx.x < En) counts[threadIdx.x] = 0;
}

// ---------------- one-time: transpose + bf16-convert weights ----------------
// w1t[e][h][d] = bf16(w1[e][d][h]) ; w2t[e][o][h] = bf16(w2[e][h][o])
__global__ __launch_bounds__(256) void prep_weights_kernel(
    const float* __restrict__ w1, const float* __restrict__ w2,
    __bf16* __restrict__ w1t, __bf16* __restrict__ w2t)
{
    const size_t i = (size_t)blockIdx.x * 256 + threadIdx.x;
    const size_t N1 = (size_t)En * Dn * Hn;
    if (i < N1) {
        const int d = (int)(i % Dn);
        const size_t r = i / Dn;
        const int h = (int)(r % Hn);
        const int e = (int)(r / Hn);
        w1t[i] = (__bf16)w1[((size_t)e * Dn + d) * Hn + h];
    } else {
        const size_t j = i - N1;
        const int h = (int)(j % Hn);
        const size_t r = j / Hn;
        const int o = (int)(r % On);
        const int e = (int)(r / On);
        w2t[j] = (__bf16)w2[((size_t)e * Hn + h) * On + o];
    }
}

// ---------------- gating: softmax -> entropy -> top-p weights + expert lists ----------------
__global__ __launch_bounds__(256) void gate_kernel(
    const float* __restrict__ x, const float* __restrict__ gw,
    const float* __restrict__ gb, float* __restrict__ weights,
    float* __restrict__ partials, int* __restrict__ lists, int* __restrict__ counts)
{
    __shared__ float gws[Dn * En];
    __shared__ float red[256];
    const int tid = threadIdx.x;
    for (int i = tid; i < Dn * En; i += 256) gws[i] = gw[i];
    __syncthreads();

    const int t = blockIdx.x * 256 + tid;
    float acc[En];
#pragma unroll
    for (int e = 0; e < En; ++e) acc[e] = gb[e];
    const float* xp = x + (size_t)t * Dn;
    for (int d = 0; d < Dn; ++d) {
        const float xv = xp[d];
#pragma unroll
        for (int e = 0; e < En; ++e) acc[e] = fmaf(xv, gws[d * En + e], acc[e]);
    }
    float m = acc[0];
#pragma unroll
    for (int e = 1; e < En; ++e) m = fmaxf(m, acc[e]);
    float p[En], s = 0.f;
#pragma unroll
    for (int e = 0; e < En; ++e) { p[e] = expf(acc[e] - m); s += p[e]; }
    const float inv = 1.f / s;
#pragma unroll
    for (int e = 0; e < En; ++e) p[e] *= inv;
    float ent = 0.f;
#pragma unroll
    for (int e = 0; e < En; ++e) ent -= p[e] * logf(p[e] + 1e-8f);

    // stable descending argsort of 8
    int ord[En];
#pragma unroll
    for (int i = 0; i < En; ++i) ord[i] = i;
    for (int i = 1; i < En; ++i) {
        const int oi = ord[i];
        const float pi = p[oi];
        int j = i - 1;
        while (j >= 0 && p[ord[j]] < pi) { ord[j + 1] = ord[j]; --j; }
        ord[j + 1] = oi;
    }
    bool mask[En];
    float cum = 0.f, msum = 0.f;
    for (int i = 0; i < En; ++i) {
        cum += p[ord[i]];
        const bool keep = (i == 0) || (cum <= TOPP);
        mask[ord[i]] = keep;
        if (keep) msum += p[ord[i]];
    }
    const float winv = 1.f / msum;
#pragma unroll
    for (int e = 0; e < En; ++e)
        weights[(size_t)t * En + e] = mask[e] ? p[e] * winv : 0.f;

    // build per-expert token lists (order nondeterministic; result order-invariant)
#pragma unroll
    for (int e = 0; e < En; ++e) {
        if (mask[e]) {
            const int pos = atomicAdd(&counts[e], 1);
            lists[(size_t)e * Tn + pos] = t;
        }
    }

    red[tid] = ent;
    __syncthreads();
    for (int sft = 128; sft > 0; sft >>= 1) {
        if (tid < sft) red[tid] += red[tid + sft];
        __syncthreads();
    }
    if (tid == 0) partials[blockIdx.x] = red[0];
}

__global__ __launch_bounds__(256) void loss_kernel(
    const float* __restrict__ partials, float* __restrict__ out_loss, int nblocks)
{
    __shared__ float red[256];
    const int tid = threadIdx.x;
    red[tid] = (tid < nblocks) ? partials[tid] : 0.f;
    __syncthreads();
    for (int sft = 128; sft > 0; sft >>= 1) {
        if (tid < sft) red[tid] += red[tid + sft];
        __syncthreads();
    }
    if (tid == 0) *out_loss = red[0] / (float)Tn;
}

// ---------------- out init: out[t,:] = sum_e w[t,e]*b2[e,:] ----------------
__global__ __launch_bounds__(256) void bias_kernel(
    const float* __restrict__ weights, const float* __restrict__ b2,
    float* __restrict__ out)
{
    const int t = blockIdx.x;
    const int n = threadIdx.x;
    float bias = 0.f;
#pragma unroll
    for (int e = 0; e < En; ++e)
        bias = fmaf(weights[(size_t)t * En + e], b2[e * On + n], bias);
    out[(size_t)t * On + n] = bias;
}

// ---------------- sparse per-expert MLP over gathered tokens ----------------
// For tokens routed to expert e: out[t,:] += w[t,e] * (gelu(x[t]@w1[e]+b1[e]) @ w2[e])
// One expert per launch (sequential) -> single writer per token -> race-free "+=".
__global__ __launch_bounds__(256) void ffn_expert_kernel(
    const float* __restrict__ x,
    const __bf16* __restrict__ w1t, const __bf16* __restrict__ w2t,
    const float* __restrict__ b1, const float* __restrict__ weights,
    const int* __restrict__ lists, const int* __restrict__ counts,
    const int e, float* __restrict__ out)
{
    __shared__ __attribute__((aligned(16))) __bf16 xs[MT * XS_S];     // 16.5 KB
    __shared__ __attribute__((aligned(16))) __bf16 wbuf[On * W2_S];   // 36 KB
    __shared__ __attribute__((aligned(16))) __bf16 hs[MT * HS_S];     // 4.5 KB
    __shared__ float b1s[Hn];
    __shared__ float wtok[MT];
    __shared__ int   toks[MT];

    const int cnt  = counts[e];
    const int base = blockIdx.x * MT;
    if (base >= cnt) return;                 // uniform early exit

    const int tid  = threadIdx.x;
    const int lane = tid & 31;
    const int wv   = tid >> 5;

    if (tid < MT) {
        int idx = base + tid;
        if (idx >= cnt) idx = cnt - 1;       // pad partial tile (stores are guarded)
        const int tok = lists[(size_t)e * Tn + idx];
        toks[tid] = tok;
        wtok[tid] = weights[(size_t)tok * En + e];
    }
    for (int i = tid; i < Hn; i += 256) b1s[i] = b1[e * Hn + i];
    __syncthreads();

    // gather x rows (f32 -> bf16), row-contiguous loads
    for (int i = tid; i < MT * Dn; i += 256) {
        const int r = i >> 8, c = i & 255;
        xs[r * XS_S + c] = (__bf16)x[(size_t)toks[r] * Dn + c];
    }

    const __bf16* w1te = w1t + (size_t)e * Hn * Dn;   // [H][D] bf16
    const __bf16* w2te = w2t + (size_t)e * On * Hn;   // [O][H] bf16
    const uint32_t wb  = (uint32_t)(uintptr_t)&wbuf[0];

    const int mi   = wv >> 2;            // 0..1 : 16-row slab
    const int ni   = wv & 3;             // 0..3 : GEMM1 h-column tile
    const int hrow = lane & 15;
    const int koff = (lane >> 4) << 3;

    v8f acc[4] = {};

    for (int hc = 0; hc < Hn / 64; ++hc) {
        __syncthreads();   // wbuf free (prev GEMM2 ds reads drained by barrier)
        // async-stage w1^T chunk: 64 rows x 512B, pure B128 copies into padded LDS rows
        for (int i = tid; i < 2048; i += 256) {
            const int n = i >> 5, c = i & 31;
            async_copy_b128(wb + (uint32_t)(n * W1_S + c * 8) * 2,
                            w1te + ((size_t)(hc * 64 + n) * Dn + c * 8));
        }
        wait_asynccnt0();
        __syncthreads();

        // ---- GEMM1: h_tile[16x16] = x[16x256] @ w1chunk[256x16] ----
        v8f c1 = {};
        const __bf16* ap = &xs[(mi * 16 + hrow) * XS_S + koff];
        const __bf16* bp = &wbuf[(ni * 16 + hrow) * W1_S + koff];
#pragma unroll
        for (int k0 = 0; k0 < Dn; k0 += 32) {
            const v16bf a = ldfrag(ap + k0);
            const v16bf b = ldfrag(bp + k0);
            c1 = __builtin_amdgcn_wmma_f32_16x16x32_bf16(
                    false, a, false, b, (short)0, c1, false, false);
        }
        {
            const int n  = lane & 15;
            const int mb = (lane >> 4) << 3;
#pragma unroll
            for (int r = 0; r < 8; ++r) {
                const int row = mi * 16 + mb + r;
                float v = c1[r] + b1s[hc * 64 + ni * 16 + n];
                v = gelu_fast(v);                    // branch-free, no EXEC divergence
                hs[row * HS_S + ni * 16 + n] = (__bf16)v;
            }
        }
        __syncthreads();   // hs complete; w1t reads finished
        // async-stage w2^T chunk: 256 rows x 128B
        for (int i = tid; i < 2048; i += 256) {
            const int n = i >> 3, c = i & 7;
            async_copy_b128(wb + (uint32_t)(n * W2_S + c * 8) * 2,
                            w2te + ((size_t)n * Hn + hc * 64 + c * 8));
        }
        wait_asynccnt0();
        __syncthreads();

        // ---- GEMM2: acc[32x256] += h[32x64] @ w2chunk[64x256] ----
        const int nb = (wv & 3) * 4;
#pragma unroll
        for (int k0 = 0; k0 < 64; k0 += 32) {
            const v16bf a = ldfrag(&hs[(mi * 16 + hrow) * HS_S + koff + k0]);
#pragma unroll
            for (int tt = 0; tt < 4; ++tt) {
                const v16bf b = ldfrag(&wbuf[((nb + tt) * 16 + hrow) * W2_S + koff + k0]);
                acc[tt] = __builtin_amdgcn_wmma_f32_16x16x32_bf16(
                            false, a, false, b, (short)0, acc[tt], false, false);
            }
        }
    }
    __syncthreads();

    // guarded scatter with routing weight applied post-GEMM2 (matches reference order):
    // out[tok,:] += w[tok,e] * acc   (single writer per token this launch)
    const int nb  = (wv & 3) * 4;
    const int nin = lane & 15;
    const int mb  = (lane >> 4) << 3;
    for (int tt = 0; tt < 4; ++tt) {
        const int n = (nb + tt) * 16 + nin;
#pragma unroll
        for (int r = 0; r < 8; ++r) {
            const int row = mi * 16 + mb + r;
            if (base + row < cnt) {
                float* po = &out[(size_t)toks[row] * On + n];
                *po = fmaf(wtok[row], acc[tt][r], *po);
            }
        }
    }
}

extern "C" void kernel_launch(void* const* d_in, const int* in_sizes, int n_in,
                              void* d_out, int out_size, void* d_ws, size_t ws_size,
                              hipStream_t stream) {
    const float* x  = (const float*)d_in[0];
    const float* gw = (const float*)d_in[1];
    const float* gb = (const float*)d_in[2];
    const float* w1 = (const float*)d_in[3];
    const float* b1 = (const float*)d_in[4];
    const float* w2 = (const float*)d_in[5];
    const float* b2 = (const float*)d_in[6];
    float* out = (float*)d_out;

    // workspace layout (16B alignment preserved for the bf16 async-B128 regions)
    float*  wsw      = (float*)d_ws;                      // [T,E] routing weights  (2 MB)
    float*  partials = wsw + (size_t)Tn * En;             // [256]
    int*    counts   = (int*)(partials + 256);            // [8]
    int*    lists    = counts + 8;                        // [E,T]                  (2 MB)
    __bf16* w1t      = (__bf16*)(lists + (size_t)En * Tn);// [E,H,D] bf16           (2 MB)
    __bf16* w2t      = w1t + (size_t)En * Hn * Dn;        // [E,O,H] bf16           (2 MB)

    init_counts_kernel<<<1, 32, 0, stream>>>(counts);
    prep_weights_kernel<<<(2 * En * Dn * Hn) / 256, 256, 0, stream>>>(w1, w2, w1t, w2t);
    gate_kernel<<<Tn / 256, 256, 0, stream>>>(x, gw, gb, wsw, partials, lists, counts);
    loss_kernel<<<1, 256, 0, stream>>>(partials, out + (size_t)Tn * On, Tn / 256);
    bias_kernel<<<Tn, 256, 0, stream>>>(wsw, b2, out);
    for (int e = 0; e < En; ++e) {
        ffn_expert_kernel<<<Tn / MT, 256, 0, stream>>>(
            x, w1t, w2t, b1, wsw, lists, counts, e, out);
    }
}
